// Attention_47313359733175
// MI455X (gfx1250) — matrix operand order repaired and checked by reading, not compiled, not run on hardware
//
#include <hip/hip_runtime.h>
#include <hip/hip_bf16.h>

typedef __attribute__((ext_vector_type(16))) _Float16 v16h;
typedef __attribute__((ext_vector_type(8)))  float    v8f;
typedef __attribute__((ext_vector_type(4)))  int      v4i_t;

#define NB 8        // batch
#define NC 128      // channels
#define NN 1024     // tokens
#define NH 8        // heads
#define NF 384      // 3*C
#define HD 144      // head dim (48*3)
#define DP 160      // padded head dim (multiple of 32)
#define QK_SCALE 0.14433756729740643f  // 48^-0.5

// CDNA5 async global->LDS path (ASYNCcnt-tracked), guarded so the file still
// compiles (with manual staging fallback) if this toolchain lacks the builtins.
#if defined(__HIP_DEVICE_COMPILE__) &&                                         \
    __has_builtin(__builtin_amdgcn_global_load_async_to_lds_b128) &&           \
    __has_builtin(__builtin_amdgcn_s_wait_asynccnt)
#define HAVE_ASYNC_LDS 1
#else
#define HAVE_ASYNC_LDS 0
#endif

// ---------------------------------------------------------------------------
// Kernel 1: QKV projection.  Y = W(384x128) * X(128x1024) per (b, component),
// scattered into per-head (B,H,N,DP) f16 layout.  SCALE folded into q.
// One 16x16 output tile per wave; 4 WMMA K=32 steps over C=128.
// ---------------------------------------------------------------------------
__global__ __launch_bounds__(256) void qkv_proj_kernel(
    const float* __restrict__ vn_x,
    const float* __restrict__ Wq, const float* __restrict__ Wk,
    const float* __restrict__ Wv,
    _Float16* __restrict__ q16, _Float16* __restrict__ k16,
    _Float16* __restrict__ v16)
{
  const int lane   = threadIdx.x & 31;
  const int wave   = threadIdx.x >> 5;
  const int r      = lane & 15;
  const int half16 = lane >> 4;

  int id = blockIdx.x * 8 + wave;
  const int ntile = id & 63;  id >>= 6;
  const int mtile = id % 24;  id /= 24;
  const int d3    = id % 3;   id /= 3;
  const int b     = id & 7;   id >>= 3;
  const int which = id;                       // 0=q 1=k 2=v

  const float* W = (which == 0) ? Wq : (which == 1) ? Wk : Wv;
  _Float16* dst  = (which == 0) ? q16 : (which == 1) ? k16 : v16;
  const float scl = (which == 0) ? QK_SCALE : 1.0f;

  const int n = ntile * 16 + r;               // B/D column = lane%16

  v8f acc = {};
  for (int kk = 0; kk < 4; ++kk) {
    // A fragment: 16x32 f16, row = lane%16, K pattern split by lane/16
    v16h a;
    const float* wrow = W + (mtile * 16 + r) * NC + kk * 32;
#pragma unroll
    for (int j = 0; j < 8; ++j) {
      const int kb = half16 * 16 + 2 * j;
      a[2 * j]     = (_Float16)wrow[kb];
      a[2 * j + 1] = (_Float16)wrow[kb + 1];
    }
    // B fragment: 32x16, col = lane%16, K rows split by lane/16
    v16h bf;
#pragma unroll
    for (int j = 0; j < 8; ++j) {
      const int c0 = kk * 32 + half16 * 16 + 2 * j;
      bf[2 * j]     = (_Float16)vn_x[((size_t)(b * NC + c0    ) * 3 + d3) * NN + n];
      bf[2 * j + 1] = (_Float16)vn_x[((size_t)(b * NC + c0 + 1) * 3 + d3) * NN + n];
    }
    acc = __builtin_amdgcn_wmma_f32_16x16x32_f16(false, a, false, bf,
                                                 (short)0, acc, false, false);
  }

  // D layout: row M = j + 8*(lane/16), col = lane%16.  o -> (head, in-head d)
#pragma unroll
  for (int j = 0; j < 8; ++j) {
    const int o  = mtile * 16 + j + 8 * half16;   // output channel in [0,384)
    const int h  = o / 48;
    const int dd = (o % 48) * 3 + d3;             // in-head dim in [0,144)
    dst[((size_t)(b * NH + h) * NN + n) * DP + dd] = (_Float16)(acc[j] * scl);
  }
}

// ---------------------------------------------------------------------------
// Kernel 2: flash attention.  One block = 128 query rows of one (b,h);
// each of the 8 waves owns 16 rows.  Loop over 32 KV blocks of 32 keys.
// S: 10 WMMAs (2 key subtiles x 5 K-steps); PV: 10 WMMAs (10 d-subtiles).
// K block staged via GLOBAL_LOAD_ASYNC_TO_LDS (when available); next block
// prefetched into L2 via global_prefetch while WMMAs run on current block.
// ---------------------------------------------------------------------------
__global__ __launch_bounds__(256) void attn_kernel(
    const _Float16* __restrict__ q16, const _Float16* __restrict__ k16,
    const _Float16* __restrict__ v16, _Float16* __restrict__ o16)
{
  __shared__ _Float16 Klds[32 * DP];       // [key][d]
  __shared__ _Float16 Vtlds[DP * 32];      // [d][key]   (transposed V)
  __shared__ _Float16 Plds[8][16 * 32];    // per-wave P staging (D->A relayout)

  const int lane   = threadIdx.x & 31;
  const int wave   = threadIdx.x >> 5;
  const int r      = lane & 15;
  const int half16 = lane >> 4;

  const int qchunk = blockIdx.x & 7;
  const int bh     = blockIdx.x >> 3;      // b*NH + h
  const int qr0    = qchunk * 128 + wave * 16;

  const _Float16* qb = q16 + (size_t)bh * NN * DP;
  const _Float16* kb = k16 + (size_t)bh * NN * DP;
  const _Float16* vb = v16 + (size_t)bh * NN * DP;

  // Preload Q A-fragments for the 5 K=32 steps over DP=160.
  v16h aq[5];
#pragma unroll
  for (int kk = 0; kk < 5; ++kk) {
    const _Float16* qrow = qb + (size_t)(qr0 + r) * DP + kk * 32 + half16 * 16;
    ((float4*)&aq[kk])[0] = *(const float4*)(qrow);        // K sub 0..7
    ((float4*)&aq[kk])[1] = *(const float4*)(qrow + 8);    // K sub 8..15
  }

  float m[8], l[8];
#pragma unroll
  for (int j = 0; j < 8; ++j) { m[j] = -1e30f; l[j] = 0.0f; }
  v8f oacc[10];
#pragma unroll
  for (int t = 0; t < 10; ++t) oacc[t] = (v8f){};

  for (int kblk = 0; kblk < 32; ++kblk) {
    // Prefetch next KV block toward L2 while this block computes.
    if (kblk + 1 < 32) {
      const size_t nxt = (size_t)((kblk + 1) * 32) * DP;
      __builtin_prefetch(kb + nxt + (size_t)threadIdx.x * 20, 0, 0);
      __builtin_prefetch(vb + nxt + (size_t)threadIdx.x * 20, 0, 0);
    }
    __syncthreads();
    // ---- Stage K block (row-major), 16B chunks (DP%8==0) ----
#if HAVE_ASYNC_LDS
    for (int idx = threadIdx.x; idx < 32 * DP / 8; idx += 256) {
      const int row = (idx * 8) / DP, col = (idx * 8) % DP;
      __builtin_amdgcn_global_load_async_to_lds_b128(
          (v4i_t*)(kb + (size_t)(kblk * 32 + row) * DP + col),
          (v4i_t*)&Klds[row * DP + col], 0, 0);
    }
#else
    for (int idx = threadIdx.x; idx < 32 * DP / 8; idx += 256) {
      const int row = (idx * 8) / DP, col = (idx * 8) % DP;
      *(float4*)&Klds[row * DP + col] =
          *(const float4*)(kb + (size_t)(kblk * 32 + row) * DP + col);
    }
#endif
    // ---- Stage V transposed: Vt[d][key] (manual; async path can't transpose)
    for (int idx = threadIdx.x; idx < 32 * DP; idx += 256) {
      const int row = idx / DP, col = idx % DP;
      Vtlds[col * 32 + row] = vb[(size_t)(kblk * 32 + row) * DP + col];
    }
#if HAVE_ASYNC_LDS
    __builtin_amdgcn_s_wait_asynccnt(0);
#endif
    __syncthreads();

    // --- S = Q * K^T (two 16x16 key subtiles) ---
    v8f s0 = {}, s1 = {};
#pragma unroll
    for (int kk = 0; kk < 5; ++kk) {
      const v16h b0 = *(const v16h*)&Klds[(      r) * DP + kk * 32 + half16 * 16];
      const v16h b1 = *(const v16h*)&Klds[(16 + r) * DP + kk * 32 + half16 * 16];
      s0 = __builtin_amdgcn_wmma_f32_16x16x32_f16(false, aq[kk], false, b0,
                                                  (short)0, s0, false, false);
      s1 = __builtin_amdgcn_wmma_f32_16x16x32_f16(false, aq[kk], false, b1,
                                                  (short)0, s1, false, false);
    }

    // --- online softmax (per row; row M = j + 8*(lane/16), 16 cols/lane grp)
#pragma unroll
    for (int j = 0; j < 8; ++j) {
      float rmax = fmaxf(s0[j], s1[j]);
#pragma unroll
      for (int off = 1; off < 16; off <<= 1)
        rmax = fmaxf(rmax, __shfl_xor(rmax, off, 32));
      const float mnew  = fmaxf(m[j], rmax);
      const float alpha = __expf(m[j] - mnew);
      const float p0 = __expf(s0[j] - mnew);
      const float p1 = __expf(s1[j] - mnew);
      float rsum = p0 + p1;
#pragma unroll
      for (int off = 1; off < 16; off <<= 1)
        rsum += __shfl_xor(rsum, off, 32);
      l[j] = l[j] * alpha + rsum;
      m[j] = mnew;
#pragma unroll
      for (int t = 0; t < 10; ++t) oacc[t][j] *= alpha;
      // stage P (D layout -> row-major 16x32 in LDS)
      Plds[wave][(j + 8 * half16) * 32 + r]      = (_Float16)p0;
      Plds[wave][(j + 8 * half16) * 32 + 16 + r] = (_Float16)p1;
    }
    __syncthreads();

    // --- O += P * V ---
    v16h ap;
    {
      const _Float16* pp = &Plds[wave][r * 32 + half16 * 16];
      ((float4*)&ap)[0] = *(const float4*)(pp);
      ((float4*)&ap)[1] = *(const float4*)(pp + 8);
    }
#pragma unroll
    for (int t = 0; t < 10; ++t) {
      const v16h bv = *(const v16h*)&Vtlds[(t * 16 + r) * 32 + half16 * 16];
      oacc[t] = __builtin_amdgcn_wmma_f32_16x16x32_f16(false, ap, false, bv,
                                                       (short)0, oacc[t],
                                                       false, false);
    }
  }

  // Epilogue: normalize and store (f16) in (bh, row, DP) layout.
  _Float16* ob = o16 + (size_t)bh * NN * DP;
#pragma unroll
  for (int j = 0; j < 8; ++j) {
    const float inv = 1.0f / l[j];
    const int row = qr0 + j + 8 * half16;
#pragma unroll
    for (int t = 0; t < 10; ++t)
      ob[(size_t)row * DP + t * 16 + r] = (_Float16)(oacc[t][j] * inv);
  }
}

// ---------------------------------------------------------------------------
// Kernel 3: output projection.  out = Wo(128x384) * X2(384x1024) per (b,d3);
// B-fragments gather the inverse head permutation from the attention output.
// ---------------------------------------------------------------------------
__global__ __launch_bounds__(256) void out_proj_kernel(
    const _Float16* __restrict__ o16, const float* __restrict__ Wo,
    float* __restrict__ out)
{
  const int lane   = threadIdx.x & 31;
  const int wave   = threadIdx.x >> 5;
  const int r      = lane & 15;
  const int half16 = lane >> 4;

  int id = blockIdx.x * 8 + wave;
  const int ntile = id & 63;  id >>= 6;
  const int mtile = id & 7;   id >>= 3;
  const int d3    = id % 3;   id /= 3;
  const int b     = id;

  const int n = ntile * 16 + r;

  v8f acc = {};
  for (int kk = 0; kk < 12; ++kk) {
    v16h a;
    const float* wrow = Wo + (mtile * 16 + r) * NF + kk * 32;
#pragma unroll
    for (int j = 0; j < 8; ++j) {
      const int kp = half16 * 16 + 2 * j;
      a[2 * j]     = (_Float16)wrow[kp];
      a[2 * j + 1] = (_Float16)wrow[kp + 1];
    }
    v16h bf;
#pragma unroll
    for (int jj = 0; jj < 16; ++jj) {
      const int o  = kk * 32 + half16 * 16 + jj;   // channel in [0,384)
      const int h  = o / 48;
      const int dd = (o % 48) * 3 + d3;
      bf[jj] = o16[((size_t)(b * NH + h) * NN + n) * DP + dd];
    }
    acc = __builtin_amdgcn_wmma_f32_16x16x32_f16(false, a, false, bf,
                                                 (short)0, acc, false, false);
  }

#pragma unroll
  for (int j = 0; j < 8; ++j) {
    const int co = mtile * 16 + j + 8 * half16;
    out[((size_t)(b * NC + co) * 3 + d3) * NN + n] = acc[j];
  }
}

// ---------------------------------------------------------------------------
extern "C" void kernel_launch(void* const* d_in, const int* in_sizes, int n_in,
                              void* d_out, int out_size, void* d_ws,
                              size_t ws_size, hipStream_t stream)
{
  (void)in_sizes; (void)n_in; (void)out_size; (void)ws_size;

  const float* vn_x = (const float*)d_in[0];
  const float* Wq   = (const float*)d_in[1];
  const float* Wk   = (const float*)d_in[2];
  const float* Wv   = (const float*)d_in[3];
  const float* Wo   = (const float*)d_in[4];
  float* out        = (float*)d_out;

  const size_t per = (size_t)NB * NH * NN * DP;   // halves per tensor
  _Float16* q16 = (_Float16*)d_ws;
  _Float16* k16 = q16 + per;
  _Float16* v16 = k16 + per;
  _Float16* o16 = v16 + per;

  // Zero workspace so padded head-dim lanes [144,160) contribute exactly 0.
  (void)hipMemsetAsync(d_ws, 0, 4 * per * sizeof(_Float16), stream);

  qkv_proj_kernel<<<(3 * NB * 3 * 24 * 64) / 8, 256, 0, stream>>>(
      vn_x, Wq, Wk, Wv, q16, k16, v16);
  attn_kernel<<<NB * NH * (NN / 128), 256, 0, stream>>>(q16, k16, v16, o16);
  out_proj_kernel<<<(NB * 3 * 8 * 64) / 8, 256, 0, stream>>>(o16, Wo, out);
}
